// EfficientMaskedAttention_12936441496277
// MI455X (gfx1250) — matrix operand order repaired and checked by reading, not compiled, not run on hardware
//
#include <hip/hip_runtime.h>
#include <hip/hip_bf16.h>

// ---------------------------------------------------------------------------
// EfficientMaskedAttention for MI455X (gfx1250, wave32, WMMA bf16)
// Pipeline:
//   prep_frag   : pre-swizzle kv_w / agg_w into WMMA A-fragment layout (bf16)
//   mask_kernel : bilinear-interp mask -> per-(b,s) 0/1 floats (257 incl cls)
//   downscale   : fused 2x2 conv + 2x2 avgpool for residual(128ch)+q(64ch)
//   attn_kernel : FUSED kv-projection (v_wmma_f32_16x16x32_bf16) + masked
//                 online-softmax attention; kv never hits HBM
//   gn_stats    : per (b,group) mean/rstd reductions
//   agg_kernel  : WMMA GEMM 128x64 with a1 groupnorm+relu fused into loader
//   final_t/out : elementwise groupnorm epilogues
// ---------------------------------------------------------------------------

typedef __attribute__((ext_vector_type(16))) __bf16 v16bf;
typedef __attribute__((ext_vector_type(8)))  float  v8f;

namespace ema {
constexpr int B    = 8;
constexpr int CCH  = 128;   // input channels
constexpr int TQ   = 400;
constexpr int TSX  = 257;
constexpr int TD   = 17;
constexpr int HID  = 64;
constexpr int OC   = 128;
constexpr int NQD  = TQ * TD;          // 6800
constexpr int SCH  = 96;               // s-chunk for fused attn
constexpr int NT   = SCH / 16;         // 6 N-tiles
constexpr int KS   = CCH / 32;         // 4 k-steps (K=128)
}

__device__ __forceinline__ unsigned short f2bf(float f) {
  unsigned int u = __float_as_uint(f);
  return (unsigned short)((u + 0x7FFFu + ((u >> 16) & 1u)) >> 16);
}
__device__ __forceinline__ float bf2f(unsigned short h) {
  return __uint_as_float(((unsigned int)h) << 16);
}

// ---------------------------------------------------------------------------
// Pre-swizzle weights into WMMA A-fragment order.
// A (16-bit, 16x32): lane L -> M = L&15 ; kbase = (L>=16)?8:0 ;
//   element e -> k = kbase + (e<8 ? e : e+8). Stored as packed u32 pairs.
// kv frags: [mt(8)][ks(4)][lane(32)][pair(8)]   agg frags: [mt(8)][ks(2)][...]
// ---------------------------------------------------------------------------
__global__ void prep_frag_kernel(const float* __restrict__ kv_w,
                                 const float* __restrict__ agg_w,
                                 unsigned int* __restrict__ kvf,
                                 unsigned int* __restrict__ agf) {
  int tid = blockIdx.x * blockDim.x + threadIdx.x;
  int stride = gridDim.x * blockDim.x;
  for (int idx = tid; idx < 8 * 4 * 32 * 8; idx += stride) {
    int j = idx & 7, lane = (idx >> 3) & 31, ks = (idx >> 8) & 3, mt = idx >> 10;
    int M = mt * 16 + (lane & 15);
    int base = (lane >= 16) ? 8 : 0;
    int e0 = 2 * j, e1 = e0 + 1;
    int k0 = ks * 32 + base + (e0 < 8 ? e0 : e0 + 8);
    int k1 = ks * 32 + base + (e1 < 8 ? e1 : e1 + 8);
    unsigned int lo = f2bf(kv_w[M * 128 + k0]);
    unsigned int hi = f2bf(kv_w[M * 128 + k1]);
    kvf[idx] = lo | (hi << 16);
  }
  for (int idx = tid; idx < 8 * 2 * 32 * 8; idx += stride) {
    int j = idx & 7, lane = (idx >> 3) & 31, ks = (idx >> 8) & 1, mt = idx >> 9;
    int M = mt * 16 + (lane & 15);
    int base = (lane >= 16) ? 8 : 0;
    int e0 = 2 * j, e1 = e0 + 1;
    int k0 = ks * 32 + base + (e0 < 8 ? e0 : e0 + 8);
    int k1 = ks * 32 + base + (e1 < 8 ? e1 : e1 + 8);
    unsigned int lo = f2bf(agg_w[M * 64 + k0]);
    unsigned int hi = f2bf(agg_w[M * 64 + k1]);
    agf[idx] = lo | (hi << 16);
  }
}

// ---------------------------------------------------------------------------
// Interpolated spatial mask -> maskf[b][257] (1.0 valid / 0.0 masked)
// ---------------------------------------------------------------------------
__global__ void mask_kernel(const int* __restrict__ s_masks,
                            float* __restrict__ maskf) {
  using namespace ema;
  int b = blockIdx.x;
  int tid = threadIdx.x;
  if (tid == 0) maskf[b * TSX + 0] = 1.f;  // cls always valid
  const float scale = 127.f / 15.f;
  int o = tid >> 4, p = tid & 15;
  float co = o * scale, cp = p * scale;
  int lo = (int)floorf(co); if (lo > 126) lo = 126;
  int lp = (int)floorf(cp); if (lp > 126) lp = 126;
  float fo = co - (float)lo, fp = cp - (float)lp;
  const int* mb = s_masks + b * 128 * 128;
  float m00 = fminf((float)mb[lo * 128 + lp], 1.f);
  float m01 = fminf((float)mb[lo * 128 + lp + 1], 1.f);
  float m10 = fminf((float)mb[(lo + 1) * 128 + lp], 1.f);
  float m11 = fminf((float)mb[(lo + 1) * 128 + lp + 1], 1.f);
  float mm = (1.f - fo) * (1.f - fp) * m00 + (1.f - fo) * fp * m01 +
             fo * (1.f - fp) * m10 + fo * fp * m11;
  maskf[b * TSX + 1 + tid] = (mm > 0.f) ? 1.f : 0.f;
}

// ---------------------------------------------------------------------------
// Fused downscale (2x2 conv folded with 2x2 avg-pool => 4-pixel pre-sums)
// o in [0,128)  -> residual path (sc_w / sc_cls_w, no bias)  -> res
// o in [128,192)-> query path   (q_w / q_cls_w, +q_b)        -> q
// One block per (b,q); corr slice staged in 32-channel LDS chunks.
// ---------------------------------------------------------------------------
__global__ __launch_bounds__(256) void downscale_kernel(
    const float* __restrict__ corr,
    const float* __restrict__ sc_w, const float* __restrict__ sc_cls_w,
    const float* __restrict__ q_w, const float* __restrict__ q_cls_w,
    const float* __restrict__ q_b,
    float* __restrict__ res, float* __restrict__ qout) {
  using namespace ema;
  const int b = blockIdx.x / TQ, qi = blockIdx.x % TQ;
  const int tid = threadIdx.x;
  constexpr int NOUT = (OC + HID) * TD;  // 3264
  __shared__ float sx[32][TSX];          // 32.9 KB
  float acc[13];
#pragma unroll
  for (int j = 0; j < 13; ++j) acc[j] = 0.f;

  for (int cc = 0; cc < 4; ++cc) {
    for (int idx = tid; idx < 32 * TSX; idx += 256) {
      int lc = idx / TSX, s = idx % TSX;
      sx[lc][s] = corr[(((long long)b * CCH + cc * 32 + lc) * TQ + qi) * TSX + s];
    }
    __syncthreads();
    for (int j = 0; j < 13; ++j) {
      int oi = tid + j * 256;
      if (oi >= NOUT) break;
      int o = oi / TD, d = oi % TD;
      const float* wcls = (o < OC) ? (sc_cls_w + o * CCH) : (q_cls_w + (o - OC) * CCH);
      const float* wsp  = (o < OC) ? (sc_w + o * CCH * 4) : (q_w + (o - OC) * CCH * 4);
      float a = acc[j];
      if (d == 0) {
        for (int lc = 0; lc < 32; ++lc) a += wcls[cc * 32 + lc] * sx[lc][0];
      } else {
        int ph = (d - 1) >> 2, pw = (d - 1) & 3;
        int base = 1 + ph * 64 + pw * 4;
        for (int lc = 0; lc < 32; ++lc) {
          const float* srow = &sx[lc][0];
          const float* w4 = wsp + (cc * 32 + lc) * 4;
#pragma unroll
          for (int i = 0; i < 2; ++i)
#pragma unroll
            for (int jj = 0; jj < 2; ++jj) {
              int p = base + i * 16 + jj;
              float s4 = srow[p] + srow[p + 2] + srow[p + 32] + srow[p + 34];
              a += w4[i * 2 + jj] * s4;
            }
        }
      }
      acc[j] = a;
    }
    __syncthreads();
  }
  for (int j = 0; j < 13; ++j) {
    int oi = tid + j * 256;
    if (oi >= NOUT) break;
    int o = oi / TD, d = oi % TD;
    float v = (d == 0) ? acc[j] : acc[j] * 0.25f;
    if (o < OC) {
      res[(((long long)b * OC + o) * TQ + qi) * TD + d] = v;
    } else {
      int oq = o - OC;
      qout[(((long long)b * HID + oq) * TQ + qi) * TD + d] = v + q_b[oq];
    }
  }
}

// ---------------------------------------------------------------------------
// FUSED kv-projection + masked attention.  One block per (b,q), 8 waves.
// s-chunks of 96: stage corr chunk bf16 -> WMMA kv chunk -> LDS bf16 ->
// 136 threads each own one (head, d) row with flash-style online softmax.
// ---------------------------------------------------------------------------
__global__ __launch_bounds__(256) void attn_kernel(
    const float* __restrict__ corr, const float* __restrict__ kv_b,
    const unsigned int* __restrict__ kvw_frag,
    const float* __restrict__ maskf, const float* __restrict__ qin,
    float* __restrict__ attn_out) {
  using namespace ema;
  const int b = blockIdx.x / TQ, qi = blockIdx.x % TQ;
  const int tid = threadIdx.x, lane = tid & 31, wv = tid >> 5;

  __shared__ unsigned short lds_x[SCH][CCH + 2];   // [s][c] bf16, 24.96 KB
  __shared__ unsigned short lds_kv[CCH][SCH + 4];  // [o][s] bf16, 25.6 KB
  __shared__ float lds_q[HID][TD];                 // 4.35 KB
  __shared__ float lds_m[SCH];                     // mask chunk
  __shared__ float lds_kvb[CCH];                   // kv bias

  // A-fragments for this wave's M-tile (mt = wv), from pre-swizzled global.
  unsigned int afr[KS][8];
#pragma unroll
  for (int ks = 0; ks < KS; ++ks) {
    const unsigned int* p = kvw_frag + (((wv * KS) + ks) * 32 + lane) * 8;
#pragma unroll
    for (int i = 0; i < 8; ++i) afr[ks][i] = p[i];
  }
  for (int idx = tid; idx < HID * TD; idx += 256) {
    int c = idx / TD, d = idx % TD;
    lds_q[c][d] = qin[(((long long)b * HID + c) * TQ + qi) * TD + d];
  }
  for (int i = tid; i < CCH; i += 256) lds_kvb[i] = kv_b[i];
  __syncthreads();

  // Online-softmax row ownership: thread r < 136 owns (head, d).
  const bool owner = tid < 8 * TD;
  const int hh = tid / TD, dd = tid % TD;
  float qreg[8];
  if (owner) {
#pragma unroll
    for (int e = 0; e < 8; ++e) qreg[e] = lds_q[hh * 8 + e][dd];
  }
  float m_run = -3.0e38f, l_run = 0.f;
  float oacc[8] = {0.f, 0.f, 0.f, 0.f, 0.f, 0.f, 0.f, 0.f};

  for (int c0 = 0; c0 < TSX; c0 += SCH) {
    // ---- phase 1: stage corr chunk (coalesced along s) + mask chunk ----
    for (int idx = tid; idx < SCH * CCH; idx += 256) {
      int c = idx / SCH, sl = idx % SCH;
      int sg = c0 + sl;
      float v = (sg < TSX)
                    ? corr[(((long long)b * CCH + c) * TQ + qi) * TSX + sg]
                    : 0.f;
      lds_x[sl][c] = f2bf(v);
    }
    for (int sl = tid; sl < SCH; sl += 256) {
      int sg = c0 + sl;
      lds_m[sl] = (sg < TSX) ? maskf[b * TSX + sg] : 0.f;
    }
    __syncthreads();

    // ---- phase 2: WMMA  kv_chunk(128 x 96) = kv_w(128x128) @ X(128x96) ----
    {
      const int n = (lane & 15);
      const int kbase = (lane >> 4) << 3;  // B frag: lanes 16-31 hold K=8-15,24-31
#pragma unroll
      for (int nt = 0; nt < NT; ++nt) {
        v8f dacc = {0.f, 0.f, 0.f, 0.f, 0.f, 0.f, 0.f, 0.f};
        const int scol = nt * 16 + n;
#pragma unroll
        for (int ks = 0; ks < KS; ++ks) {
          union { unsigned int u[8]; v16bf v; } bfr, af;
          const unsigned int* bp =
              (const unsigned int*)&lds_x[scol][ks * 32 + kbase];
#pragma unroll
          for (int i = 0; i < 8; ++i) bfr.u[i] = bp[(i & 3) + ((i >> 2) << 3)];
#pragma unroll
          for (int i = 0; i < 8; ++i) af.u[i] = afr[ks][i];
          dacc = __builtin_amdgcn_wmma_f32_16x16x32_bf16(
              false, af.v, false, bfr.v, (short)0, dacc, false, false);
        }
        const int rbase = wv * 16 + ((lane >> 4) << 3);
#pragma unroll
        for (int r = 0; r < 8; ++r) {
          int o = rbase + r;
          lds_kv[o][scol] = f2bf(dacc[r] + lds_kvb[o]);
        }
      }
    }
    __syncthreads();

    // ---- phase 3: online masked softmax update (k rows 0..63, v 64..127) ---
    if (owner) {
      int wlim = (TSX - c0 < SCH) ? (TSX - c0) : SCH;
      for (int sl = 0; sl < wlim; ++sl) {
        float msk = lds_m[sl];
        if (msk == 0.f) continue;
        float x = 0.f;
#pragma unroll
        for (int e = 0; e < 8; ++e)
          x += qreg[e] * bf2f(lds_kv[hh * 8 + e][sl]);
        float m_new = fmaxf(m_run, x);
        float scl = __expf(m_run - m_new);
        float p = __expf(x - m_new);
        l_run = l_run * scl + p;
#pragma unroll
        for (int e = 0; e < 8; ++e)
          oacc[e] = oacc[e] * scl + p * bf2f(lds_kv[HID + hh * 8 + e][sl]);
        m_run = m_new;
      }
    }
    __syncthreads();
  }

  if (owner) {
    float inv = 1.f / l_run;
#pragma unroll
    for (int e = 0; e < 8; ++e)
      attn_out[(((long long)b * HID + hh * 8 + e) * TQ + qi) * TD + dd] =
          oacc[e] * inv;
  }
}

// ---------------------------------------------------------------------------
// Group-norm statistics: one block per (b, group); channels are contiguous.
// ---------------------------------------------------------------------------
__global__ __launch_bounds__(256) void gn_stats_kernel(
    const float* __restrict__ src, float* __restrict__ stats, int Cn) {
  using namespace ema;
  int b = blockIdx.x >> 2, g = blockIdx.x & 3;
  int cpg = Cn >> 2;
  int count = cpg * NQD;
  const float* base = src + ((long long)b * Cn + (long long)g * cpg) * NQD;
  float s = 0.f, ss = 0.f;
  for (int idx = threadIdx.x; idx < count; idx += 256) {
    float x = base[idx];
    s += x;
    ss += x * x;
  }
  __shared__ float rs[256], rq[256];
  rs[threadIdx.x] = s;
  rq[threadIdx.x] = ss;
  __syncthreads();
  for (int off = 128; off > 0; off >>= 1) {
    if (threadIdx.x < off) {
      rs[threadIdx.x] += rs[threadIdx.x + off];
      rq[threadIdx.x] += rq[threadIdx.x + off];
    }
    __syncthreads();
  }
  if (threadIdx.x == 0) {
    float mean = rs[0] / (float)count;
    float var = rq[0] / (float)count - mean * mean;
    stats[(b * 4 + g) * 2 + 0] = mean;
    stats[(b * 4 + g) * 2 + 1] = rsqrtf(var + 1e-5f);
  }
}

// ---------------------------------------------------------------------------
// agg GEMM (M=128, K=64) with a1 groupnorm+relu fused into the LDS loader.
// ---------------------------------------------------------------------------
__global__ __launch_bounds__(256) void agg_kernel(
    const float* __restrict__ a1, const unsigned int* __restrict__ aggw_frag,
    const float* __restrict__ st_a1, const float* __restrict__ a1_g,
    const float* __restrict__ a1_b, float* __restrict__ aggo) {
  using namespace ema;
  constexpr int NTB = (NQD + 127) / 128;  // 54
  const int b = blockIdx.x / NTB;
  const int n0 = (blockIdx.x % NTB) * 128;
  const int tid = threadIdx.x, lane = tid & 31, wv = tid >> 5;
  __shared__ unsigned short lds_x[128][66];  // [n][c] bf16, 16.9 KB

  for (int idx = tid; idx < 128 * 64; idx += 256) {
    int c = idx >> 7, nl = idx & 127;
    int n = n0 + nl;
    float v = 0.f;
    if (n < NQD) {
      float x = a1[((long long)b * HID + c) * NQD + n];
      int g = c >> 4;
      float mean = st_a1[(b * 4 + g) * 2 + 0];
      float rstd = st_a1[(b * 4 + g) * 2 + 1];
      v = fmaxf((x - mean) * rstd * a1_g[c] + a1_b[c], 0.f);
    }
    lds_x[nl][c] = f2bf(v);
  }
  unsigned int afr[2][8];
#pragma unroll
  for (int ks = 0; ks < 2; ++ks) {
    const unsigned int* p = aggw_frag + ((wv * 2 + ks) * 32 + lane) * 8;
#pragma unroll
    for (int i = 0; i < 8; ++i) afr[ks][i] = p[i];
  }
  __syncthreads();

  const int nl = (lane & 15);
  const int kbase = (lane >> 4) << 3;
#pragma unroll
  for (int nt = 0; nt < 8; ++nt) {
    v8f dacc = {0.f, 0.f, 0.f, 0.f, 0.f, 0.f, 0.f, 0.f};
    const int nloc = nt * 16 + nl;
#pragma unroll
    for (int ks = 0; ks < 2; ++ks) {
      union { unsigned int u[8]; v16bf v; } bfr, af;
      const unsigned int* bp = (const unsigned int*)&lds_x[nloc][ks * 32 + kbase];
#pragma unroll
      for (int i = 0; i < 8; ++i) bfr.u[i] = bp[(i & 3) + ((i >> 2) << 3)];
#pragma unroll
      for (int i = 0; i < 8; ++i) af.u[i] = afr[ks][i];
      dacc = __builtin_amdgcn_wmma_f32_16x16x32_bf16(
          false, af.v, false, bfr.v, (short)0, dacc, false, false);
    }
    int scol = n0 + nloc;
    int rbase = wv * 16 + ((lane >> 4) << 3);
    if (scol < NQD) {
#pragma unroll
      for (int r = 0; r < 8; ++r)
        aggo[((long long)b * OC + rbase + r) * NQD + scol] = dacc[r];
    }
  }
}

// ---------------------------------------------------------------------------
// t = relu(gn_a2(agg)) + relu(gn_sc(res))
// ---------------------------------------------------------------------------
__global__ void final_t_kernel(const float* __restrict__ agg,
                               const float* __restrict__ res,
                               const float* __restrict__ st_a2,
                               const float* __restrict__ st_sc,
                               const float* __restrict__ a2_g,
                               const float* __restrict__ a2_b,
                               const float* __restrict__ sc_g,
                               const float* __restrict__ sc_b,
                               float* __restrict__ t) {
  using namespace ema;
  const long long total = (long long)B * OC * NQD;
  for (long long idx = blockIdx.x * 256LL + threadIdx.x; idx < total;
       idx += (long long)gridDim.x * 256) {
    int c = (int)((idx / NQD) & 127);
    int b = (int)(idx / ((long long)OC * NQD));
    int g = c >> 5;
    float ma = st_a2[(b * 4 + g) * 2], ra = st_a2[(b * 4 + g) * 2 + 1];
    float ms = st_sc[(b * 4 + g) * 2], rs = st_sc[(b * 4 + g) * 2 + 1];
    float va = fmaxf((agg[idx] - ma) * ra * a2_g[c] + a2_b[c], 0.f);
    float vs = fmaxf((res[idx] - ms) * rs * sc_g[c] + sc_b[c], 0.f);
    t[idx] = va + vs;
  }
}

__global__ void final_out_kernel(const float* __restrict__ t,
                                 const float* __restrict__ st_on,
                                 const float* __restrict__ on_g,
                                 const float* __restrict__ on_b,
                                 float* __restrict__ out) {
  using namespace ema;
  const long long total = (long long)B * OC * NQD;
  for (long long idx = blockIdx.x * 256LL + threadIdx.x; idx < total;
       idx += (long long)gridDim.x * 256) {
    int c = (int)((idx / NQD) & 127);
    int b = (int)(idx / ((long long)OC * NQD));
    int g = c >> 5;
    float m = st_on[(b * 4 + g) * 2], r = st_on[(b * 4 + g) * 2 + 1];
    out[idx] = (t[idx] - m) * r * on_g[c] + on_b[c];
  }
}

// ---------------------------------------------------------------------------
extern "C" void kernel_launch(void* const* d_in, const int* in_sizes, int n_in,
                              void* d_out, int out_size, void* d_ws,
                              size_t ws_size, hipStream_t stream) {
  using namespace ema;
  const float* corr     = (const float*)d_in[0];
  const int*   s_masks  = (const int*)d_in[1];
  const float* sc_w     = (const float*)d_in[2];
  const float* sc_cls_w = (const float*)d_in[3];
  const float* sc_g     = (const float*)d_in[4];
  const float* sc_b     = (const float*)d_in[5];
  const float* q_w      = (const float*)d_in[6];
  const float* q_cls_w  = (const float*)d_in[7];
  const float* q_b      = (const float*)d_in[8];
  const float* kv_w     = (const float*)d_in[9];
  const float* kv_b     = (const float*)d_in[10];
  const float* a1_g     = (const float*)d_in[11];
  const float* a1_b     = (const float*)d_in[12];
  const float* agg_w    = (const float*)d_in[13];
  const float* a2_g     = (const float*)d_in[14];
  const float* a2_b     = (const float*)d_in[15];
  const float* on_g     = (const float*)d_in[16];
  const float* on_b     = (const float*)d_in[17];
  float* out = (float*)d_out;

  // Workspace layout (floats). t reuses [q | attn] region (both dead by then).
  float* w = (float*)d_ws;
  const long long QSZ = (long long)B * HID * NQD;  // 3,481,600
  const long long RSZ = (long long)B * OC * NQD;   // 6,963,200
  float* ws_q    = w;
  float* ws_attn = w + QSZ;
  float* ws_res  = w + 2 * QSZ;
  float* ws_agg  = w + 2 * QSZ + RSZ;
  float* ws_t    = w;  // overwrites q+attn (exactly RSZ floats)
  float* ws_mask = w + 2 * QSZ + 2 * RSZ;            // B*257
  float* st_sc   = ws_mask + B * TSX;                // 64 each
  float* st_a1   = st_sc + 64;
  float* st_a2   = st_sc + 128;
  float* st_on   = st_sc + 192;
  unsigned int* kvw_frag  = (unsigned int*)(st_sc + 256);
  unsigned int* aggw_frag = kvw_frag + 8 * 4 * 32 * 8;

  prep_frag_kernel<<<16, 256, 0, stream>>>(kv_w, agg_w, kvw_frag, aggw_frag);
  mask_kernel<<<B, 256, 0, stream>>>(s_masks, ws_mask);
  downscale_kernel<<<B * TQ, 256, 0, stream>>>(corr, sc_w, sc_cls_w, q_w,
                                               q_cls_w, q_b, ws_res, ws_q);
  attn_kernel<<<B * TQ, 256, 0, stream>>>(corr, kv_b, kvw_frag, ws_mask, ws_q,
                                          ws_attn);
  gn_stats_kernel<<<B * 4, 256, 0, stream>>>(ws_attn, st_a1, HID);
  gn_stats_kernel<<<B * 4, 256, 0, stream>>>(ws_res, st_sc, OC);
  agg_kernel<<<B * ((NQD + 127) / 128), 256, 0, stream>>>(
      ws_attn, aggw_frag, st_a1, a1_g, a1_b, ws_agg);
  gn_stats_kernel<<<B * 4, 256, 0, stream>>>(ws_agg, st_a2, OC);
  final_t_kernel<<<27200, 256, 0, stream>>>(ws_agg, ws_res, st_a2, st_sc, a2_g,
                                            a2_b, sc_g, sc_b, ws_t);
  gn_stats_kernel<<<B * 4, 256, 0, stream>>>(ws_t, st_on, OC);
  final_out_kernel<<<27200, 256, 0, stream>>>(ws_t, st_on, on_g, on_b, out);
}
